// PFGAT_214748365437
// MI455X (gfx1250) — compile-verified, matrix operand order checked by reading, not run on hardware
//
#include <hip/hip_runtime.h>

// ---------------------------------------------------------------------------
// CDNA5 (gfx1250, wave32) implementation of the spatiotemporal transformer +
// GAT reference.  Heavy math runs on v_wmma_f32_16x16x32_bf16.
// Model dims (fixed by reference):
#define BB 2
#define TT 16
#define NN 2048
#define FIN 24
#define FPAD 32
#define HH 128
#define EE 32768
#define RTOT (BB*TT*NN)     // 65536 rows (b,t,n)
#define RL   (BB*NN)        // 4096 rows (b,n)
#define ETOT (BB*EE + BB*NN)// 69632 edges incl self loops
#define MBY  ((size_t)1 << 20)
// ---------------------------------------------------------------------------

typedef __attribute__((ext_vector_type(16))) __bf16 v16bf;
typedef __attribute__((ext_vector_type(8)))  float  v8f;

union Frag { uint4 u[2]; unsigned short s[16]; v16bf v; };
static_assert(sizeof(Frag) == 32, "frag size");

__device__ inline unsigned short f2bfu(float f) {
    unsigned u = __float_as_uint(f);
    unsigned r = u + 0x7FFFu + ((u >> 16) & 1u);   // round-to-nearest-even
    return (unsigned short)(r >> 16);
}
__device__ inline float bfu2f(unsigned short s) {
    return __uint_as_float(((unsigned)s) << 16);
}
// monotonic float<->uint for atomicMax-based segment max
__device__ inline unsigned fenc(float f) {
    unsigned u = __float_as_uint(f);
    return (u & 0x80000000u) ? ~u : (u | 0x80000000u);
}
__device__ inline float fdec(unsigned u) {
    unsigned v = (u & 0x80000000u) ? (u & 0x7FFFFFFFu) : ~u;
    return __uint_as_float(v);
}

#define WMMA_BF16(a, b, c) __builtin_amdgcn_wmma_f32_16x16x32_bf16( \
    false, (a), false, (b), (short)0, (c), false, false)

// Load a 16x32 bf16 WMMA fragment (A-layout; identical layout serves B when the
// source is stored "n-major, k-contiguous", i.e. transposed weights Wt[N][K]).
// element e of lane l holds M[row0 + (l&15)][kb + (e<8 ? e : e+8) + 8*(l>>4)]
__device__ inline v16bf ldfrag(const unsigned short* __restrict__ base, int ld,
                               int row0, int kb) {
    int lane = threadIdx.x & 31, half = lane >> 4, r = lane & 15;
    const unsigned short* p = base + (size_t)(row0 + r) * ld + kb + half * 8;
    Frag f;
    f.u[0] = *(const uint4*)p;
    f.u[1] = *(const uint4*)(p + 16);
    return f.v;
}

// ---------------------------------------------------------------------------
// Generic bf16 WMMA GEMM: C[M,N] = act(Amap @ Wt^T + bias + res + emb)
// A row mapping: arow = ((row/aCH)*aMUL + aADD)*aCH + row%aCH  (identity: 1,1,0)
// (used to gather the t=T-1 rows without a copy).  trChunk!=0 stores bf16
// output transposed per chunk:  out[(row/ch)*N*ch + col*ch + row%ch]  (V^T).
// Block: 256 threads = 8 waves; wave tile 16x64; block tile 128 x 64.
// K-loop is software-pipelined: fragments for iteration kb+32 are issued
// before the WMMAs of iteration kb, so waits are partial, not loadcnt==0.
// ---------------------------------------------------------------------------
__global__ __launch_bounds__(256) void gemm_bf16(
    const unsigned short* __restrict__ A, const unsigned short* __restrict__ Wt,
    int M, int Ncols, int K,
    int aCH, int aMUL, int aADD,
    const float* __restrict__ bias,
    const float* __restrict__ res, int rCH, int rMUL, int rADD,
    const float* __restrict__ emb, int embMod,
    int act,
    float* __restrict__ outF, unsigned short* __restrict__ outB,
    int trChunk)
{
    int wave = threadIdx.x >> 5;
    int lane = threadIdx.x & 31, half = lane >> 4, l16 = lane & 15;
    int mrow = blockIdx.x * 128 + wave * 16;
    int n0b  = blockIdx.y * 64;
    if (mrow >= M) return;

    int arowbase = ((mrow / aCH) * aMUL + aADD) * aCH + mrow % aCH;

    v8f z = {0.f,0.f,0.f,0.f,0.f,0.f,0.f,0.f};
    v8f acc[4] = {z, z, z, z};

    // prologue loads
    v16bf a  = ldfrag(A, K, arowbase, 0);
    v16bf b0 = ldfrag(Wt, K, n0b,      0);
    v16bf b1 = ldfrag(Wt, K, n0b + 16, 0);
    v16bf b2 = ldfrag(Wt, K, n0b + 32, 0);
    v16bf b3 = ldfrag(Wt, K, n0b + 48, 0);
    for (int kb = 32; kb < K; kb += 32) {
        v16bf a2 = ldfrag(A, K, arowbase, kb);
        v16bf c0 = ldfrag(Wt, K, n0b,      kb);
        v16bf c1 = ldfrag(Wt, K, n0b + 16, kb);
        v16bf c2 = ldfrag(Wt, K, n0b + 32, kb);
        v16bf c3 = ldfrag(Wt, K, n0b + 48, kb);
        acc[0] = WMMA_BF16(a, b0, acc[0]);
        acc[1] = WMMA_BF16(a, b1, acc[1]);
        acc[2] = WMMA_BF16(a, b2, acc[2]);
        acc[3] = WMMA_BF16(a, b3, acc[3]);
        a = a2; b0 = c0; b1 = c1; b2 = c2; b3 = c3;
    }
    acc[0] = WMMA_BF16(a, b0, acc[0]);
    acc[1] = WMMA_BF16(a, b1, acc[1]);
    acc[2] = WMMA_BF16(a, b2, acc[2]);
    acc[3] = WMMA_BF16(a, b3, acc[3]);

    int rrowbase = res ? ((mrow / rCH) * rMUL + rADD) * rCH + mrow % rCH : 0;
#pragma unroll
    for (int j = 0; j < 4; j++) {
        int n0 = n0b + j * 16 + l16;
        float bv = bias ? bias[n0] : 0.f;
#pragma unroll
        for (int r8 = 0; r8 < 8; r8++) {
            int row = mrow + r8 + 8 * half;
            float v = acc[j][r8] + bv;
            if (res) v += res[(size_t)(rrowbase + r8 + 8 * half) * Ncols + n0];
            if (emb) v += emb[(size_t)(row % embMod) * Ncols + n0];
            if (act == 1) v = v > 0.f ? v : 0.f;
            size_t o = trChunk
                ? ((size_t)(row / trChunk) * Ncols * trChunk + (size_t)n0 * trChunk + row % trChunk)
                : ((size_t)row * Ncols + n0);
            if (outF) outF[o] = v;
            if (outB) outB[o] = f2bfu(v);
        }
    }
}

// ---------------------------------------------------------------------------
// Flash attention, causal over node axis.  One (b,t) per blockIdx.y.
// Wave handles 16 queries; keys in blocks of 32.  Computes S^T = K @ Q^T so
// that the exp(P) values sit in-lane exactly where the next WMMA's B-fragment
// needs them (no LDS transpose); softmax reduction is one shfl_xor(16).
// O accumulated as O^T = V^T @ P^T (V pre-transposed [h][n]).
// All 8 K-fragments are loaded before the S WMMA chain; the 8 V-fragments
// are issued right after it so they overlap the softmax VALU work.
// ---------------------------------------------------------------------------
__global__ __launch_bounds__(256) void flash_attn(
    const unsigned short* __restrict__ q,   // [bt][n][h] bf16
    const unsigned short* __restrict__ k,   // [bt][n][h] bf16
    const unsigned short* __restrict__ vt,  // [bt][h][n] bf16
    unsigned short* __restrict__ o)         // [bt][n][h] bf16
{
    int bt = blockIdx.y;
    int wave = threadIdx.x >> 5, lane = threadIdx.x & 31;
    int half = lane >> 4, l16 = lane & 15;
    int q0 = blockIdx.x * 128 + wave * 16;

    const unsigned short* Qb = q  + (size_t)bt * NN * HH;
    const unsigned short* Kb = k  + (size_t)bt * NN * HH;
    const unsigned short* Vb = vt + (size_t)bt * HH * NN;

    v16bf bq[4];
#pragma unroll
    for (int j = 0; j < 4; j++) bq[j] = ldfrag(Qb, HH, q0, j * 32);

    v8f z = {0.f,0.f,0.f,0.f,0.f,0.f,0.f,0.f};
    v8f oacc[8] = {z, z, z, z, z, z, z, z};
    float mrun = -3.0e38f, lrun = 0.f;
    const float scale = 0.08838834764831845f;  // 1/sqrt(128)
    int qcol = q0 + l16;

    for (int kb = 0; kb <= q0 + 15; kb += 32) {
        // ---- S^T = K @ Q^T : load all 8 K frags, then 8 WMMAs ----
        v16bf ak[8];
#pragma unroll
        for (int s = 0; s < 2; s++)
#pragma unroll
            for (int j = 0; j < 4; j++)
                ak[s * 4 + j] = ldfrag(Kb, HH, kb + 16 * s, j * 32);
        v8f st[2] = {z, z};
#pragma unroll
        for (int s = 0; s < 2; s++)
#pragma unroll
            for (int j = 0; j < 4; j++)
                st[s] = WMMA_BF16(ak[s * 4 + j], bq[j], st[s]);

        // issue V^T fragment loads now: independent of st, overlaps softmax
        v16bf av[8];
#pragma unroll
        for (int t = 0; t < 8; t++) av[t] = ldfrag(Vb, NN, t * 16, kb);
        // prefetch next key block (gfx1250 global_prefetch_b8)
        if (kb + 32 <= q0 + 15) {
            __builtin_prefetch(Kb + (size_t)(kb + 32) * HH, 0, 0);
            __builtin_prefetch(Vb + kb + 32, 0, 0);
        }

        // ---- online softmax in C layout (rows = keys, lanes = queries) ----
        float p0[8], p1[8];
        float mloc = -3.0e38f;
#pragma unroll
        for (int r = 0; r < 8; r++) {
            float s0 = st[0][r] * scale;
            float s1 = st[1][r] * scale;
            int key0 = kb + r + 8 * half;
            if (key0 > qcol)      s0 = -3.0e38f;   // causal: key <= query
            if (key0 + 16 > qcol) s1 = -3.0e38f;
            p0[r] = s0; p1[r] = s1;
            mloc = fmaxf(mloc, fmaxf(s0, s1));
        }
        mloc = fmaxf(mloc, __shfl_xor(mloc, 16));
        float mnew = fmaxf(mrun, mloc);
        float corr = __expf(mrun - mnew);
        float rs = 0.f;
        Frag pf;
#pragma unroll
        for (int r = 0; r < 8; r++) {
            float e0 = __expf(p0[r] - mnew), e1 = __expf(p1[r] - mnew);
            rs += e0 + e1;
            pf.s[r]     = f2bfu(e0);   // key = kb + r + 8*half       (e=0..7)
            pf.s[r + 8] = f2bfu(e1);   // key = kb + 16 + r + 8*half  (e=8..15)
        }
        rs += __shfl_xor(rs, 16);
        lrun = lrun * corr + rs;
        mrun = mnew;
#pragma unroll
        for (int t = 0; t < 8; t++) oacc[t] = oacc[t] * corr;
        // ---- O^T += V^T @ P^T ----
#pragma unroll
        for (int t = 0; t < 8; t++) oacc[t] = WMMA_BF16(av[t], pf.v, oacc[t]);
    }
    float inv = 1.f / lrun;
    unsigned short* orow = o + ((size_t)bt * NN + q0 + l16) * HH;
#pragma unroll
    for (int t = 0; t < 8; t++) {
        union { unsigned short s[8]; uint4 u; } tmp;
#pragma unroll
        for (int r = 0; r < 8; r++) tmp.s[r] = f2bfu(oacc[t][r] * inv);
        *(uint4*)(orow + t * 16 + half * 8) = tmp.u;
    }
}

// --------------------------- small kernels ---------------------------------
// weight -> bf16 transposed [N][Kpad] (K padded with zeros)
__global__ void wconv(const float* __restrict__ W, unsigned short* __restrict__ Wt,
                      int K, int Nn, int Kpad) {
    int idx = blockIdx.x * 256 + threadIdx.x;
    if (idx >= Nn * Kpad) return;
    int n = idx / Kpad, kk = idx % Kpad;
    Wt[idx] = (kk < K) ? f2bfu(W[(size_t)kk * Nn + n]) : (unsigned short)0;
}

// concat(x,ERA5) -> raw bf16 [r][32] and LN(24) bf16 [r][32]; one wave per row
__global__ __launch_bounds__(256) void prep_xin(
    const float* __restrict__ x, const float* __restrict__ era,
    const float* __restrict__ g, const float* __restrict__ bta,
    unsigned short* __restrict__ xraw, unsigned short* __restrict__ xln) {
    int wave = threadIdx.x >> 5, lane = threadIdx.x & 31;
    int row = blockIdx.x * 8 + wave;
    if (row >= RTOT) return;
    float v = 0.f;
    if (lane < 8)       v = x[(size_t)row * 8 + lane];
    else if (lane < 24) v = era[(size_t)row * 16 + lane - 8];
    float s = v;
    for (int off = 16; off; off >>= 1) s += __shfl_xor(s, off);
    float mu = s * (1.f / 24.f);
    float d = (lane < 24) ? v - mu : 0.f;
    float qq = d * d;
    for (int off = 16; off; off >>= 1) qq += __shfl_xor(qq, off);
    float rstd = rsqrtf(qq * (1.f / 24.f) + 1e-5f);
    float gg = (lane < 24) ? g[lane] : 0.f;
    float bb = (lane < 24) ? bta[lane] : 0.f;
    xraw[(size_t)row * 32 + lane] = (lane < 24) ? f2bfu(v) : (unsigned short)0;
    xln [(size_t)row * 32 + lane] = (lane < 24) ? f2bfu(d * rstd * gg + bb) : (unsigned short)0;
}

// LayerNorm width 128, one block per row, f32 in -> bf16 out
__global__ __launch_bounds__(128) void ln128(
    const float* __restrict__ xin, const float* __restrict__ g,
    const float* __restrict__ b, unsigned short* __restrict__ out) {
    int row = blockIdx.x, tid = threadIdx.x;
    float v = xin[(size_t)row * 128 + tid];
    __shared__ float sm[4], sq[4];
    float s = v;
    for (int off = 16; off; off >>= 1) s += __shfl_xor(s, off);
    if ((tid & 31) == 0) sm[tid >> 5] = s;
    __syncthreads();
    float mu = (sm[0] + sm[1] + sm[2] + sm[3]) * (1.f / 128.f);
    float d = v - mu, qq = d * d;
    for (int off = 16; off; off >>= 1) qq += __shfl_xor(qq, off);
    if ((tid & 31) == 0) sq[tid >> 5] = qq;
    __syncthreads();
    float var = (sq[0] + sq[1] + sq[2] + sq[3]) * (1.f / 128.f);
    out[(size_t)row * 128 + tid] = f2bfu(d * rsqrtf(var + 1e-5f) * g[tid] + b[tid]);
}

// attn_axis2 collapsed to the only needed output row (t = T-1): 16-way
// matvec softmax per (b,n).  One block of 128 threads per node.
__global__ __launch_bounds__(128) void attn2(
    const float* __restrict__ q2,            // [RL][128] f32 (t=15 rows)
    const unsigned short* __restrict__ k2,   // [RTOT][128] bf16
    const unsigned short* __restrict__ v2,   // [RTOT][128] bf16
    unsigned short* __restrict__ o2) {       // [RL][128] bf16
    int bn = blockIdx.x, b = bn >> 11, n = bn & 2047;
    int tid = threadIdx.x, wave = tid >> 5, lane = tid & 31;
    __shared__ float ss[16], sa[16];
    float qv[4];
#pragma unroll
    for (int i = 0; i < 4; i++) qv[i] = q2[(size_t)bn * 128 + lane + 32 * i];
#pragma unroll
    for (int i = 0; i < 4; i++) {
        int t = wave * 4 + i;
        size_t r = ((size_t)(b * 16 + t) * 2048 + n) * 128;
        float p = 0.f;
#pragma unroll
        for (int jj = 0; jj < 4; jj++) p += qv[jj] * bfu2f(k2[r + lane + 32 * jj]);
        for (int off = 16; off; off >>= 1) p += __shfl_xor(p, off);
        if (lane == 0) ss[t] = p * 0.08838834764831845f;
    }
    __syncthreads();
    if (tid == 0) {
        float m = -3e38f;
        for (int t = 0; t < 16; t++) m = fmaxf(m, ss[t]);
        float sum = 0.f;
        for (int t = 0; t < 16; t++) { float e = __expf(ss[t] - m); sa[t] = e; sum += e; }
        float inv = 1.f / sum;
        for (int t = 0; t < 16; t++) sa[t] *= inv;
    }
    __syncthreads();
    float acc = 0.f;
#pragma unroll
    for (int t = 0; t < 16; t++) {
        size_t r = ((size_t)(b * 16 + t) * 2048 + n) * 128;
        acc += sa[t] * bfu2f(v2[r + tid]);
    }
    o2[(size_t)bn * 128 + tid] = f2bfu(acc);
}

// ----------------------------- GAT -----------------------------------------
__device__ inline void edge_sd(const int* __restrict__ ei, int e, int& s, int& d) {
    if (e < BB * EE) {
        int smp = e >> 15, i = e & (EE - 1);
        s = ei[i] + smp * NN;
        d = ei[EE + i] + smp * NN;
    } else {
        int j = e - BB * EE; s = j; d = j;
    }
}

__global__ __launch_bounds__(128) void gat_coef(
    const float* __restrict__ h, const float* __restrict__ asrcW,
    const float* __restrict__ adstW, float* __restrict__ asrc,
    float* __restrict__ adst, int heads, int C) {
    int m = blockIdx.x, wave = threadIdx.x >> 5, lane = threadIdx.x & 31;
    if (wave >= heads) return;
    float s = 0.f, d = 0.f;
    for (int c = lane; c < C; c += 32) {
        float hv = h[(size_t)m * heads * C + wave * C + c];
        s += hv * asrcW[wave * C + c];
        d += hv * adstW[wave * C + c];
    }
    for (int off = 16; off; off >>= 1) { s += __shfl_xor(s, off); d += __shfl_xor(d, off); }
    if (lane == 0) { asrc[m * heads + wave] = s; adst[m * heads + wave] = d; }
}

__global__ void edge_alpha(const int* __restrict__ ei, const float* __restrict__ asrc,
                           const float* __restrict__ adst, float* __restrict__ alpha,
                           unsigned* __restrict__ amax, int heads) {
    int idx = blockIdx.x * 256 + threadIdx.x;
    int e = idx / heads, hd = idx % heads;
    if (e >= ETOT) return;
    int s, d; edge_sd(ei, e, s, d);
    float a = asrc[s * heads + hd] + adst[d * heads + hd];
    a = a > 0.f ? a : 0.2f * a;                 // leaky_relu(0.2)
    alpha[idx] = a;
    atomicMax(&amax[d * heads + hd], fenc(a));
}

__global__ void edge_exp(const int* __restrict__ ei, float* __restrict__ alpha,
                         const unsigned* __restrict__ amax, float* __restrict__ denom,
                         int heads) {
    int idx = blockIdx.x * 256 + threadIdx.x;
    int e = idx / heads, hd = idx % heads;
    if (e >= ETOT) return;
    int s, d; edge_sd(ei, e, s, d);
    float ea = __expf(alpha[idx] - fdec(amax[d * heads + hd]));
    alpha[idx] = ea;
    atomicAdd(&denom[d * heads + hd], ea);
}

__global__ void edge_scatter(const int* __restrict__ ei, const float* __restrict__ alpha,
                             const float* __restrict__ denom, const float* __restrict__ h,
                             float* __restrict__ out, int heads, int C) {
    int F = heads * C;
    long idx = (long)blockIdx.x * 256 + threadIdx.x;
    long e = idx / F; int f = (int)(idx % F);
    if (e >= ETOT) return;
    int hd = f / C;
    int s, d; edge_sd(ei, (int)e, s, d);
    float w = alpha[e * heads + hd] / (denom[d * heads + hd] + 1e-16f);
    atomicAdd(&out[(size_t)d * F + f], h[(size_t)s * F + f] * w);
}

__global__ void elu_bias(const float* __restrict__ in, const float* __restrict__ bias,
                         unsigned short* __restrict__ outB, long n, int F) {
    long i = (long)blockIdx.x * 256 + threadIdx.x;
    if (i >= n) return;
    float v = in[i] + bias[i % F];
    v = v > 0.f ? v : (__expf(v) - 1.f);        // elu
    outB[i] = f2bfu(v);
}

__global__ void fill_u32(unsigned* p, unsigned v, long n) {
    long i = (long)blockIdx.x * 256 + threadIdx.x;
    if (i < n) p[i] = v;
}
__global__ void fill_f32(float* p, float v, long n) {
    long i = (long)blockIdx.x * 256 + threadIdx.x;
    if (i < n) p[i] = v;
}

__global__ __launch_bounds__(128) void final_out(
    const float* __restrict__ o2acc, const float* __restrict__ g2b,
    const float* __restrict__ fw, const float* __restrict__ fb,
    float* __restrict__ out) {
    int node = blockIdx.x * 4 + (threadIdx.x >> 5);
    int lane = threadIdx.x & 31;
    if (node >= RL) return;
    float a = 0.f;
    for (int c = lane; c < 64; c += 32) a += (o2acc[(size_t)node * 64 + c] + g2b[c]) * fw[c];
    for (int off = 16; off; off >>= 1) a += __shfl_xor(a, off);
    if (lane == 0) out[node] = a + fb[0];
}

// ---------------------------------------------------------------------------
extern "C" void kernel_launch(void* const* d_in, const int* in_sizes, int n_in,
                              void* d_out, int out_size, void* d_ws, size_t ws_size,
                              hipStream_t stream) {
    (void)in_sizes; (void)n_in; (void)out_size; (void)ws_size;
    // inputs in setup_inputs() insertion order (params recursively)
    const float* x    = (const float*)d_in[0];
    const float* era  = (const float*)d_in[1];
    const int*   ei   = (const int*)d_in[2];
    const float* emb  = (const float*)d_in[3];
    const float* skipW= (const float*)d_in[4];
    const float* skipb= (const float*)d_in[5];
    const float* ln1g = (const float*)d_in[6];
    const float* ln1b = (const float*)d_in[7];
    const float* tWq  = (const float*)d_in[8];
    const float* tbq  = (const float*)d_in[9];
    const float* tWk  = (const float*)d_in[10];
    const float* tbk  = (const float*)d_in[11];
    const float* tWv  = (const float*)d_in[12];
    const float* tbv  = (const float*)d_in[13];
    const float* tWo  = (const float*)d_in[14];
    const float* tbo  = (const float*)d_in[15];
    const float* ln2g = (const float*)d_in[16];
    const float* ln2b = (const float*)d_in[17];
    const float* sWq  = (const float*)d_in[18];
    const float* sbq  = (const float*)d_in[19];
    const float* sWk  = (const float*)d_in[20];
    const float* sbk  = (const float*)d_in[21];
    const float* sWv  = (const float*)d_in[22];
    const float* sbv  = (const float*)d_in[23];
    const float* sWo  = (const float*)d_in[24];
    const float* sbo  = (const float*)d_in[25];
    const float* ln3g = (const float*)d_in[26];
    const float* ln3b = (const float*)d_in[27];
    const float* fW1  = (const float*)d_in[28];
    const float* fb1  = (const float*)d_in[29];
    const float* fW2  = (const float*)d_in[30];
    const float* fb2  = (const float*)d_in[31];
    const float* g1W  = (const float*)d_in[32];
    const float* g1as = (const float*)d_in[33];
    const float* g1ad = (const float*)d_in[34];
    const float* g1b  = (const float*)d_in[35];
    const float* g2W  = (const float*)d_in[36];
    const float* g2as = (const float*)d_in[37];
    const float* g2ad = (const float*)d_in[38];
    const float* g2b  = (const float*)d_in[39];
    const float* ffnW = (const float*)d_in[40];
    const float* ffnb = (const float*)d_in[41];
    float* out = (float*)d_out;

    // ---- workspace (aliased; ~120 MB peak, fits L2) ----
    char* wsb = (char*)d_ws;
    size_t off = 0;
    auto alloc = [&](size_t bytes) -> void* {
        void* p = wsb + off;
        off = (off + bytes + 255) & ~(size_t)255;
        return p;
    };
    typedef unsigned short u16;
    float* h1 = (float*)alloc((size_t)RTOT * HH * 4);  // 32 MB, live to stage 3
    u16* qT   = (u16*)alloc((size_t)RTOT * HH * 2);    // 16 MB
    u16* kT   = (u16*)alloc((size_t)RTOT * HH * 2);    // 16 MB
    u16* vT   = (u16*)alloc((size_t)RTOT * HH * 2);    // 16 MB  [bt][h][n]
    u16* S1   = (u16*)alloc((size_t)RTOT * HH * 2);    // 16 MB  o_bf, later ln2o
    char* P   = (char*)alloc(24 * MBY);                // 24 MB reuse pool
    // pool phase 1 (inputs -> xin); dead after the 4 stage-1 GEMMs
    u16* xin_raw = (u16*)(P + 0);                      // 4 MB
    u16* xin_ln  = (u16*)(P + 4 * MBY);                // 4 MB
    // pool phase >=3 (safe: producers launch after last readers of phase 1)
    float* q2f = (float*)(P + 0);                      // 2 MB
    u16* o2bf  = (u16*)(P + 2 * MBY);                  // 1 MB
    float* h2  = (float*)(P + 3 * MBY);                // 2 MB
    u16* ln3o  = (u16*)(P + 5 * MBY);                  // 1 MB
    u16* ff1o  = (u16*)(P + 6 * MBY);                  // 2 MB
    u16* xg    = (u16*)(P + 8 * MBY);                  // 1 MB
    float* hg1 = (float*)(P + 9 * MBY);                // 2 MB
    float* as1 = (float*)(P + 11 * MBY);               // 64 KB
    float* ad1 = (float*)(P + 11 * MBY + (64 << 10));
    unsigned* am1 = (unsigned*)(P + 11 * MBY + (128 << 10));
    float* dn1 = (float*)(P + 11 * MBY + (192 << 10));
    float* al1 = (float*)(P + 11 * MBY + (256 << 10)); // 1.07 MB
    float* o1a = (float*)(P + 13 * MBY);               // 2 MB
    u16* xg2   = (u16*)(P + 15 * MBY);                 // 1 MB
    float* hg2 = (float*)(P + 16 * MBY);               // 1 MB
    float* as2 = (float*)(P + 17 * MBY);
    float* ad2 = (float*)(P + 17 * MBY + (64 << 10));
    unsigned* am2 = (unsigned*)(P + 17 * MBY + (128 << 10));
    float* dn2 = (float*)(P + 17 * MBY + (192 << 10));
    float* al2 = (float*)(P + 17 * MBY + (256 << 10)); // 0.27 MB
    float* o2a = (float*)(P + 18 * MBY);               // 1 MB
    // aliases of large phase-1/2 buffers
    u16* o_bf = S1;       // flash output; dead after Wo GEMM
    u16* ln2o = S1;       // written by ln128 after Wo GEMM reads o_bf
    u16* k2   = qT;       // qT dead after flash
    u16* v2   = kT;       // kT dead after flash
    // bf16 transposed weights [N][Kpad]
    u16* skipWT = (u16*)alloc(128 * 32 * 2);
    u16* tWqT = (u16*)alloc(128 * 32 * 2);
    u16* tWkT = (u16*)alloc(128 * 32 * 2);
    u16* tWvT = (u16*)alloc(128 * 32 * 2);
    u16* tWoT = (u16*)alloc(128 * 128 * 2);
    u16* sWqT = (u16*)alloc(128 * 128 * 2);
    u16* sWkT = (u16*)alloc(128 * 128 * 2);
    u16* sWvT = (u16*)alloc(128 * 128 * 2);
    u16* sWoT = (u16*)alloc(128 * 128 * 2);
    u16* fW1T = (u16*)alloc(256 * 128 * 2);
    u16* fW2T = (u16*)alloc(128 * 256 * 2);
    u16* g1WT = (u16*)alloc(128 * 128 * 2);
    u16* g2WT = (u16*)alloc(64 * 128 * 2);

    auto WC = [&](const float* W, u16* Wt, int K, int Nn, int Kpad) {
        int n = Nn * Kpad;
        wconv<<<(n + 255) / 256, 256, 0, stream>>>(W, Wt, K, Nn, Kpad);
    };
    WC(skipW, skipWT, 24, 128, 32);
    WC(tWq, tWqT, 24, 128, 32);  WC(tWk, tWkT, 24, 128, 32);  WC(tWv, tWvT, 24, 128, 32);
    WC(tWo, tWoT, 128, 128, 128);
    WC(sWq, sWqT, 128, 128, 128); WC(sWk, sWkT, 128, 128, 128);
    WC(sWv, sWvT, 128, 128, 128); WC(sWo, sWoT, 128, 128, 128);
    WC(fW1, fW1T, 128, 256, 128); WC(fW2, fW2T, 256, 128, 256);
    WC(g1W, g1WT, 128, 128, 128); WC(g2W, g2WT, 128, 64, 128);

    // ---- stage 1: inputs -> xin (raw + LN over F=24) ----
    prep_xin<<<RTOT / 8, 256, 0, stream>>>(x, era, ln1g, ln1b, xin_raw, xin_ln);

#define GEMM(A_,W_,M_,N_,K_,aC,aM,aA,bi,re,rC,rM,rA,em,eM,ac,oF,oB,tr) \
    gemm_bf16<<<dim3((M_)/128,(N_)/64), 256, 0, stream>>>( \
        A_, W_, M_, N_, K_, aC, aM, aA, bi, re, rC, rM, rA, em, eM, ac, oF, oB, tr)
    // q,k,v projections (bf16 out), skip path (f32 into h1)
    GEMM(xin_ln, tWqT, RTOT, 128, 32, 1,1,0, tbq, nullptr,1,1,0, nullptr,1, 0, nullptr, qT, 0);
    GEMM(xin_ln, tWkT, RTOT, 128, 32, 1,1,0, tbk, nullptr,1,1,0, nullptr,1, 0, nullptr, kT, 0);
    GEMM(xin_ln, tWvT, RTOT, 128, 32, 1,1,0, tbv, nullptr,1,1,0, nullptr,1, 0, nullptr, vT, NN); // V^T store
    GEMM(xin_raw, skipWT, RTOT, 128, 32, 1,1,0, skipb, nullptr,1,1,0, nullptr,1, 0, h1, nullptr, 0);

    // ---- stage 2: 32x causal flash attention over nodes ----
    flash_attn<<<dim3(NN / 128, BB * TT), 256, 0, stream>>>(qT, kT, vT, o_bf);

    // h1 = skip + attn @ Wo + bo   (in place on h1)
    GEMM(o_bf, tWoT, RTOT, 128, 128, 1,1,0, tbo, h1,1,1,0, nullptr,1, 0, h1, nullptr, 0);

    // ---- stage 3: attn2 (only t=T-1 output row needed) ----
    ln128<<<RTOT, 128, 0, stream>>>(h1, ln2g, ln2b, ln2o);
    GEMM(ln2o, sWkT, RTOT, 128, 128, 1,1,0, sbk, nullptr,1,1,0, nullptr,1, 0, nullptr, k2, 0);
    GEMM(ln2o, sWvT, RTOT, 128, 128, 1,1,0, sbv, nullptr,1,1,0, nullptr,1, 0, nullptr, v2, 0);
    // Q gathered at t=15 via A row mapping: arow = ((row/2048)*16+15)*2048 + row%2048
    GEMM(ln2o, sWqT, RL, 128, 128, 2048,16,15, sbq, nullptr,1,1,0, nullptr,1, 0, q2f, nullptr, 0);
    attn2<<<RL, 128, 0, stream>>>(q2f, k2, v2, o2bf);
    // h2 = h1[t=15] + o2 @ Wo + bo
    GEMM(o2bf, sWoT, RL, 128, 128, 1,1,0, sbo, h1,2048,16,15, nullptr,1, 0, h2, nullptr, 0);

    // ---- stage 4: FF + emb -> xg ----
    ln128<<<RL, 128, 0, stream>>>(h2, ln3g, ln3b, ln3o);
    GEMM(ln3o, fW1T, RL, 256, 128, 1,1,0, fb1, nullptr,1,1,0, nullptr,1, 1 /*relu*/, nullptr, ff1o, 0);
    GEMM(ff1o, fW2T, RL, 128, 256, 1,1,0, fb2, h2,1,1,0, emb,2048, 0, nullptr, xg, 0);

    // ---- stage 5: GAT layer 1 (heads=4, C=32) ----
    GEMM(xg, g1WT, RL, 128, 128, 1,1,0, nullptr, nullptr,1,1,0, nullptr,1, 0, hg1, nullptr, 0);
    gat_coef<<<RL, 128, 0, stream>>>(hg1, g1as, g1ad, as1, ad1, 4, 32);
    fill_u32<<<(RL * 4 + 255) / 256, 256, 0, stream>>>(am1, 0x007FFFFFu, (long)RL * 4); // enc(-inf)
    fill_f32<<<(RL * 4 + 255) / 256, 256, 0, stream>>>(dn1, 0.f, (long)RL * 4);
    fill_f32<<<(RL * 128 + 255) / 256, 256, 0, stream>>>(o1a, 0.f, (long)RL * 128);
    edge_alpha<<<(ETOT * 4 + 255) / 256, 256, 0, stream>>>(ei, as1, ad1, al1, am1, 4);
    edge_exp<<<(ETOT * 4 + 255) / 256, 256, 0, stream>>>(ei, al1, am1, dn1, 4);
    edge_scatter<<<(int)(((long)ETOT * 128 + 255) / 256), 256, 0, stream>>>(ei, al1, dn1, hg1, o1a, 4, 32);
    elu_bias<<<(RL * 128 + 255) / 256, 256, 0, stream>>>(o1a, g1b, xg2, (long)RL * 128, 128);

    // ---- stage 6: GAT layer 2 (heads=1, C=64) ----
    GEMM(xg2, g2WT, RL, 64, 128, 1,1,0, nullptr, nullptr,1,1,0, nullptr,1, 0, hg2, nullptr, 0);
    gat_coef<<<RL, 128, 0, stream>>>(hg2, g2as, g2ad, as2, ad2, 1, 64);
    fill_u32<<<(RL + 255) / 256, 256, 0, stream>>>(am2, 0x007FFFFFu, (long)RL);
    fill_f32<<<(RL + 255) / 256, 256, 0, stream>>>(dn2, 0.f, (long)RL);
    fill_f32<<<(RL * 64 + 255) / 256, 256, 0, stream>>>(o2a, 0.f, (long)RL * 64);
    edge_alpha<<<(ETOT + 255) / 256, 256, 0, stream>>>(ei, as2, ad2, al2, am2, 1);
    edge_exp<<<(ETOT + 255) / 256, 256, 0, stream>>>(ei, al2, am2, dn2, 1);
    edge_scatter<<<(int)(((long)ETOT * 64 + 255) / 256), 256, 0, stream>>>(ei, al2, dn2, hg2, o2a, 1, 64);

    // ---- final 64->1 projection ----
    final_out<<<RL / 4, 128, 0, stream>>>(o2a, g2b, ffnW, ffnb, out);
#undef GEMM
}
